// RNNModel_77171972374636
// MI455X (gfx1250) — matrix operand order repaired and checked by reading, not compiled, not run on hardware
//
#include <hip/hip_runtime.h>
#include <hip/hip_bf16.h>

// ---------------------------------------------------------------------------
// Problem constants (from reference): V=32000, D=512, H=512, B=16, S=257
// ---------------------------------------------------------------------------
#define CV 32000
#define CD 512
#define CH 512
#define CB 16
#define CS 257
#define CT 256            // S-1 timesteps
#define CM (CB*CT)        // 4096 rows for the big GEMMs
#define G4H (4*CH)        // 2048 gate columns

typedef __attribute__((ext_vector_type(16))) __bf16 v16bf;
typedef __attribute__((ext_vector_type(8)))  __bf16 v8bf;
typedef __attribute__((ext_vector_type(8)))  float  v8f;
typedef __attribute__((ext_vector_type(4)))  int    v4i;

__device__ inline v8f v8f_zero() {
    v8f z = {0.f,0.f,0.f,0.f,0.f,0.f,0.f,0.f};
    return z;
}

__device__ inline v16bf frag2(const __bf16* p0, const __bf16* p1) {
    v8bf lo = *(const v8bf*)p0;
    v8bf hi = *(const v8bf*)p1;
    v16bf r;
#pragma unroll
    for (int i = 0; i < 8; ++i) { r[i] = lo[i]; r[i + 8] = hi[i]; }
    return r;
}

__device__ inline v8f wmma_bf16(v16bf a, v16bf b, v8f c) {
    return __builtin_amdgcn_wmma_f32_16x16x32_bf16(false, a, false, b,
                                                   (short)0, c, false, false);
}

__device__ inline float sigm(float x) { return 1.f / (1.f + __expf(-x)); }

// ---------------------------------------------------------------------------
// CDNA5 async LDS staging: GLOBAL_LOAD_ASYNC_TO_LDS_B128 (ASYNCcnt-tracked).
// Builtin signature (from hipcc diagnostic): arg0 = v4i* in global (__device__)
// address space, arg1 = LDS side, then offset + cpol immediates.
// ---------------------------------------------------------------------------
typedef __attribute__((address_space(1))) v4i* gas_v4i;
typedef __attribute__((address_space(3))) v4i* las_v4i;

__device__ inline void async_copy16(const void* gptr, void* ldsptr) {
#if defined(__has_builtin) && __has_builtin(__builtin_amdgcn_global_load_async_to_lds_b128)
    __builtin_amdgcn_global_load_async_to_lds_b128(
        (gas_v4i)gptr, (las_v4i)ldsptr, 0, 0);
#else
    unsigned lds = (unsigned)(uintptr_t)ldsptr;
    asm volatile("global_load_async_to_lds_b128 %0, %1, off"
                 :: "v"(lds), "v"(gptr) : "memory");
#endif
}

__device__ inline void async_wait(int n) {  // called with constants 0 / 3
#if defined(__has_builtin) && __has_builtin(__builtin_amdgcn_s_wait_asynccnt)
    if (n == 0) __builtin_amdgcn_s_wait_asynccnt(0);
    else        __builtin_amdgcn_s_wait_asynccnt(3);
#else
    if (n == 0) asm volatile("s_wait_asynccnt 0x0" ::: "memory");
    else        asm volatile("s_wait_asynccnt 0x3" ::: "memory");
#endif
}

// ---------------------------------------------------------------------------
// Generic f32-input bf16-WMMA GEMM: C[M,N] = A[M,K] * B + bias.
//  flags bit0: B stored (N,K) row-major    bit1: tanh
//  flags bit2: bias present                bit3: store C as bf16
// Block 256 thr = 8 waves, tile 64x128, K step 32. Exact-multiple sizes only.
// ---------------------------------------------------------------------------
#define BM 64
#define BN 128
#define BKK 32

__global__ __launch_bounds__(256) void gemm_bf16_kernel(
    const float* __restrict__ A, const float* __restrict__ Bm,
    const float* __restrict__ bias, void* __restrict__ Cv,
    int N, int K, long sA, long sB, long sC, int flags)
{
    __shared__ __bf16 As[BM][BKK];
    __shared__ __bf16 Bs[BN][BKK];

    const int tid  = threadIdx.x;
    const int wave = tid >> 5;
    const int lane = tid & 31;
    const int half = lane >> 4;
    const int l16  = lane & 15;
    const int wm   = wave >> 2;
    const int wn   = wave & 3;

    const long m0 = (long)blockIdx.x * BM;
    const long n0 = (long)blockIdx.y * BN;
    A  += (long)blockIdx.z * sA;
    Bm += (long)blockIdx.z * sB;

    v8f acc[2][2];
#pragma unroll
    for (int i = 0; i < 2; ++i)
#pragma unroll
        for (int j = 0; j < 2; ++j) acc[i][j] = v8f_zero();

    const int ar  = tid >> 2;
    const int ac  = (tid & 3) * 8;
    const int btr = tid >> 1;
    const int btc = (tid & 1) * 16;
    const int bkr = tid >> 3;
    const int bnc = (tid & 7) * 16;

    for (int k0 = 0; k0 < K; k0 += BKK) {
        __syncthreads();
        {
            const float* s = A + (m0 + ar) * (long)K + k0 + ac;
#pragma unroll
            for (int i = 0; i < 8; ++i) As[ar][ac + i] = (__bf16)s[i];
            if (k0 + BKK < K) __builtin_prefetch(s + BKK, 0, 1);
        }
        if (flags & 1) {
            const float* s = Bm + (n0 + btr) * (long)K + k0 + btc;
#pragma unroll
            for (int i = 0; i < 16; ++i) Bs[btr][btc + i] = (__bf16)s[i];
            if (k0 + BKK < K) __builtin_prefetch(s + BKK, 0, 1);
        } else {
            const float* s = Bm + (long)(k0 + bkr) * N + n0 + bnc;
#pragma unroll
            for (int i = 0; i < 16; ++i) Bs[bnc + i][bkr] = (__bf16)s[i];
            if (k0 + BKK < K) __builtin_prefetch(s + (long)BKK * N, 0, 1);
        }
        __syncthreads();

#pragma unroll
        for (int tm = 0; tm < 2; ++tm) {
            const __bf16* ap = &As[wm * 32 + tm * 16 + l16][0];
            v16bf a = frag2(ap + half * 16, ap + half * 16 + 8);
#pragma unroll
            for (int tn = 0; tn < 2; ++tn) {
                const __bf16* bp = &Bs[wn * 32 + tn * 16 + l16][half * 16];
                v16bf b = frag2(bp, bp + 8);
                acc[tm][tn] = wmma_bf16(a, b, acc[tm][tn]);
            }
        }
    }

    float*  Cf = (float*) Cv + (long)blockIdx.z * sC;
    __bf16* Cb = (__bf16*)Cv + (long)blockIdx.z * sC;
#pragma unroll
    for (int tm = 0; tm < 2; ++tm)
#pragma unroll
        for (int tn = 0; tn < 2; ++tn) {
            long n = n0 + wn * 32 + tn * 16 + l16;
            float bv = (flags & 4) ? bias[n] : 0.f;
#pragma unroll
            for (int r = 0; r < 8; ++r) {
                long m = m0 + wm * 32 + tm * 16 + half * 8 + r;
                float v = acc[tm][tn][r] + bv;
                if (flags & 2) v = tanhf(v);
                if (flags & 8) Cb[m * (long)N + n] = (__bf16)v;
                else           Cf[m * (long)N + n] = v;
            }
        }
}

// ---------------------------------------------------------------------------
// bf16-input WMMA GEMM with double-buffered ASYNC LDS staging.
// A: (M,K) bf16 row-major.  B: (N,K) bf16 row-major (pre-transposed).
//  flags bit1: tanh   bit2: bias   bit3: store bf16
// ---------------------------------------------------------------------------
__global__ __launch_bounds__(256) void gemm_bf16x_kernel(
    const __bf16* __restrict__ A, const __bf16* __restrict__ Bm,
    const float* __restrict__ bias, void* __restrict__ Cv,
    int N, int K, long sA, long sB, long sC, int flags)
{
    __shared__ __bf16 As[2][BM][BKK];   // 2 x 4 KB
    __shared__ __bf16 Bs[2][BN][BKK];   // 2 x 8 KB

    const int tid  = threadIdx.x;
    const int wave = tid >> 5;
    const int lane = tid & 31;
    const int half = lane >> 4;
    const int l16  = lane & 15;
    const int wm   = wave >> 2;
    const int wn   = wave & 3;

    const long m0 = (long)blockIdx.x * BM;
    const long n0 = (long)blockIdx.y * BN;
    A  += (long)blockIdx.z * sA;
    Bm += (long)blockIdx.z * sB;

    v8f acc[2][2];
#pragma unroll
    for (int i = 0; i < 2; ++i)
#pragma unroll
        for (int j = 0; j < 2; ++j) acc[i][j] = v8f_zero();

    // 16B chunks: A tile = 256 chunks (1/thread), B tile = 512 chunks (2/thread)
    const int a_row = tid >> 2;
    const int a_c   = (tid & 3) * 8;
    const int b_row0 = tid >> 2;                  // rows 0..63
    const int b_c0   = (tid & 3) * 8;
    const int b_row1 = 64 + (tid >> 2);           // rows 64..127
    const int b_c1   = (tid & 3) * 8;

    auto issue = [&](int buf, int k0) {
        async_copy16(A  + (m0 + a_row)  * (long)K + k0 + a_c,  &As[buf][a_row][a_c]);
        async_copy16(Bm + (n0 + b_row0) * (long)K + k0 + b_c0, &Bs[buf][b_row0][b_c0]);
        async_copy16(Bm + (n0 + b_row1) * (long)K + k0 + b_c1, &Bs[buf][b_row1][b_c1]);
    };

    const int nk = K / BKK;
    issue(0, 0);
    int buf = 0;
    for (int kt = 0; kt < nk; ++kt) {
        const bool more = (kt + 1) < nk;
        if (more) issue(buf ^ 1, (kt + 1) * BKK);   // overlap next tile
        async_wait(more ? 3 : 0);                    // current tile landed
        __syncthreads();

#pragma unroll
        for (int tm = 0; tm < 2; ++tm) {
            const __bf16* ap = &As[buf][wm * 32 + tm * 16 + l16][0];
            v16bf a = frag2(ap + half * 16, ap + half * 16 + 8);
#pragma unroll
            for (int tn = 0; tn < 2; ++tn) {
                const __bf16* bp = &Bs[buf][wn * 32 + tn * 16 + l16][half * 16];
                v16bf b = frag2(bp, bp + 8);
                acc[tm][tn] = wmma_bf16(a, b, acc[tm][tn]);
            }
        }
        __syncthreads();
        buf ^= 1;
    }

    float*  Cf = (float*) Cv + (long)blockIdx.z * sC;
    __bf16* Cb = (__bf16*)Cv + (long)blockIdx.z * sC;
#pragma unroll
    for (int tm = 0; tm < 2; ++tm)
#pragma unroll
        for (int tn = 0; tn < 2; ++tn) {
            long n = n0 + wn * 32 + tn * 16 + l16;
            float bv = (flags & 4) ? bias[n] : 0.f;
#pragma unroll
            for (int r = 0; r < 8; ++r) {
                long m = m0 + wm * 32 + tm * 16 + half * 8 + r;
                float v = acc[tm][tn][r] + bv;
                if (flags & 2) v = tanhf(v);
                if (flags & 8) Cb[m * (long)N + n] = (__bf16)v;
                else           Cf[m * (long)N + n] = v;
            }
        }
}

// ---------------------------------------------------------------------------
// Embedding gather: seq[t][b][:] = emb[tok[b][t]][:]   ((T,B,D) layout)
// ---------------------------------------------------------------------------
__global__ void embed_kernel(const int* __restrict__ tok,
                             const float* __restrict__ emb,
                             float* __restrict__ seq)
{
    int t = blockIdx.x, b = blockIdx.y;
    int id = tok[b * CS + t];
    const float* s = emb + (long)id * CD;
    float* d = seq + ((long)t * CB + b) * CD;
    for (int i = threadIdx.x; i < CD; i += blockDim.x) d[i] = s[i];
}

// ---------------------------------------------------------------------------
// Coalesced 32x32 tiled transpose + f32 -> bf16: dst[n][k] = src[k][n].
// grid (K/32, N/32), 256 threads.
// ---------------------------------------------------------------------------
__global__ __launch_bounds__(256) void transpose_bf16_kernel(
    const float* __restrict__ src, __bf16* __restrict__ dst, int K, int N)
{
    __shared__ float tile[32][33];
    const int k0 = blockIdx.x * 32, n0 = blockIdx.y * 32;
    const int r = threadIdx.x >> 5;        // 0..7
    const int c = threadIdx.x & 31;        // 0..31
#pragma unroll
    for (int rr = r; rr < 32; rr += 8)
        tile[rr][c] = src[(long)(k0 + rr) * N + n0 + c];
    __syncthreads();
#pragma unroll
    for (int rr = r; rr < 32; rr += 8)
        dst[(long)(n0 + rr) * K + k0 + c] = (__bf16)tile[c][rr];
}

// ---------------------------------------------------------------------------
// Zero h ping-pong buffers + global-barrier counter (every replay).
// ---------------------------------------------------------------------------
__global__ void init_kernel(float* h0, float* h1, unsigned* ctr)
{
    for (int i = threadIdx.x; i < 2 * CB * CH; i += blockDim.x) {
        h0[i] = 0.f; h1[i] = 0.f;
    }
    if (threadIdx.x == 0) *ctr = 0u;
}

// ---------------------------------------------------------------------------
// Persistent LSTM recurrence: 32 blocks x 128 threads (4 waves = 4 gates).
// ---------------------------------------------------------------------------
#define RNB 32

__device__ inline void gsync(unsigned* ctr, unsigned target)
{
    __syncthreads();
    __threadfence();
    if (threadIdx.x == 0) {
        atomicAdd(ctr, 1u);
        while (__hip_atomic_load(ctr, __ATOMIC_RELAXED,
                                 __HIP_MEMORY_SCOPE_AGENT) < target) {
            __builtin_amdgcn_s_sleep(8);
        }
    }
    __syncthreads();
    __threadfence();
}

__device__ inline v8f gemm_chain(const __bf16* hA, const __bf16* wB,
                                 int half, v8f acc)
{
#pragma unroll
    for (int kb = 0; kb < 16; ++kb) {
        int k0 = kb * 32;
        v16bf a = frag2(hA + k0 + half * 16, hA + k0 + half * 16 + 8);
        v16bf b = frag2(wB + k0 + half * 16, wB + k0 + half * 16 + 8);
        acc = wmma_bf16(a, b, acc);
    }
    return acc;
}

__global__ __launch_bounds__(128) void lstm_recur_kernel(
    const float* __restrict__ Xg,
    const __bf16* __restrict__ W0T,
    const __bf16* __restrict__ W1iT,
    const __bf16* __restrict__ W1hT,
    const float* __restrict__ b1,
    float* __restrict__ h0, float* __restrict__ h1,
    float* __restrict__ out, __bf16* __restrict__ outbf,
    unsigned* __restrict__ ctr)
{
    __shared__ __bf16 h0l[CB * CH];
    __shared__ __bf16 h1l[CB * CH];
    __shared__ float  gbuf[4 * 256];
    __shared__ float  cl[2 * 256];

    const int tid  = threadIdx.x;
    const int wave = tid >> 5;
    const int lane = tid & 31;
    const int half = lane >> 4;
    const int l16  = lane & 15;
    const int bs   = blockIdx.x;

    const int gcol = wave * CH + bs * 16 + l16;
    const __bf16* w0p  = W0T  + (long)gcol * CH;
    const __bf16* w1ip = W1iT + (long)gcol * CH;
    const __bf16* w1hp = W1hT + (long)gcol * CH;
    const float bias1 = b1[gcol];

    for (int e = tid; e < 512; e += 128) cl[e] = 0.f;
    unsigned target = 0;
    __syncthreads();

    for (int t = 0; t < CT; ++t) {
        const float* h0rd = h0 + (t & 1) * (CB * CH);
        float*       h0wr = h0 + ((t + 1) & 1) * (CB * CH);
        const float* h1rd = h1 + (t & 1) * (CB * CH);
        float*       h1wr = h1 + ((t + 1) & 1) * (CB * CH);

        // ----- layer 0 -----
        for (int e = tid; e < CB * CH; e += 128) h0l[e] = (__bf16)h0rd[e];
        __syncthreads();

        v8f acc;
        {
            const float* xrow = Xg + ((long)t * CB) * G4H + gcol;
#pragma unroll
            for (int r = 0; r < 8; ++r) acc[r] = xrow[(long)(half * 8 + r) * G4H];
        }
        acc = gemm_chain(h0l + l16 * CH, w0p, half, acc);
#pragma unroll
        for (int r = 0; r < 8; ++r)
            gbuf[wave * 256 + (half * 8 + r) * 16 + l16] = acc[r];
        __syncthreads();

        for (int e = tid; e < 256; e += 128) {
            int b = e >> 4, j = e & 15;
            float iv = gbuf[0 * 256 + e], fv = gbuf[1 * 256 + e];
            float gv = gbuf[2 * 256 + e], ov = gbuf[3 * 256 + e];
            float c = cl[e];
            c = sigm(fv) * c + sigm(iv) * tanhf(gv);
            float h = sigm(ov) * tanhf(c);
            cl[e] = c;
            h0wr[b * CH + bs * 16 + j] = h;
        }
        gsync(ctr, (target += RNB));

        // ----- layer 1 -----
        for (int e = tid; e < CB * CH; e += 128) {
            h0l[e] = (__bf16)h0wr[e];
            h1l[e] = (__bf16)h1rd[e];
        }
        __syncthreads();

#pragma unroll
        for (int r = 0; r < 8; ++r) acc[r] = bias1;
        acc = gemm_chain(h0l + l16 * CH, w1ip, half, acc);
        acc = gemm_chain(h1l + l16 * CH, w1hp, half, acc);
#pragma unroll
        for (int r = 0; r < 8; ++r)
            gbuf[wave * 256 + (half * 8 + r) * 16 + l16] = acc[r];
        __syncthreads();

        for (int e = tid; e < 256; e += 128) {
            int b = e >> 4, j = e & 15;
            float iv = gbuf[0 * 256 + e], fv = gbuf[1 * 256 + e];
            float gv = gbuf[2 * 256 + e], ov = gbuf[3 * 256 + e];
            float c = cl[256 + e];
            c = sigm(fv) * c + sigm(iv) * tanhf(gv);
            float h = sigm(ov) * tanhf(c);
            cl[256 + e] = c;
            h1wr[b * CH + bs * 16 + j] = h;
            long oidx = ((long)b * CT + t) * CH + bs * 16 + j;
            out[oidx]   = h;
            outbf[oidx] = (__bf16)h;
        }
        gsync(ctr, (target += RNB));
    }
}

// ---------------------------------------------------------------------------
// Masked column softmax over i (axis=1), written transposed: wT[b][j][i].
// ---------------------------------------------------------------------------
__global__ void attn_softmax_kernel(const float* __restrict__ scores,
                                    const int* __restrict__ tok,
                                    float* __restrict__ wT)
{
    int j = blockIdx.x, b = blockIdx.y, i = threadIdx.x;
    __shared__ float red[256];

    bool m = tok[b * CS + i + 1] != 0;
    float s = m ? scores[((long)(b * CT + i)) * CT + j] : -1e30f;
    red[i] = s; __syncthreads();
    for (int off = 128; off > 0; off >>= 1) {
        if (i < off) red[i] = fmaxf(red[i], red[i + off]);
        __syncthreads();
    }
    float mx = red[0]; __syncthreads();
    float e = m ? __expf(s - mx) : 0.f;
    red[i] = e; __syncthreads();
    for (int off = 128; off > 0; off >>= 1) {
        if (i < off) red[i] += red[i + off];
        __syncthreads();
    }
    float sum = red[0];
    float w = (sum > 0.f) ? e / sum : 0.f;
    wT[((long)(b * CT + j)) * CT + i] = w;
}

__global__ void concat_kernel(const float* __restrict__ ctx,
                              const float* __restrict__ out,
                              float* __restrict__ cat)
{
    long m = blockIdx.x;
    const float* c = ctx + m * CH;
    const float* o = out + m * CH;
    float* d = cat + m * (2 * CH);
    for (int i = threadIdx.x; i < CH; i += blockDim.x) {
        d[i] = c[i];
        d[CH + i] = o[i];
    }
}

// ---------------------------------------------------------------------------
// Host-side launch
// ---------------------------------------------------------------------------
extern "C" void kernel_launch(void* const* d_in, const int* in_sizes, int n_in,
                              void* d_out, int out_size, void* d_ws, size_t ws_size,
                              hipStream_t stream)
{
    (void)in_sizes; (void)n_in; (void)out_size; (void)ws_size;

    const int*   tok  = (const int*)  d_in[0];
    const float* emb  = (const float*)d_in[1];
    const float* Wih0 = (const float*)d_in[2];
    const float* Whh0 = (const float*)d_in[3];
    const float* b0   = (const float*)d_in[4];
    const float* Wih1 = (const float*)d_in[5];
    const float* Whh1 = (const float*)d_in[6];
    const float* b1   = (const float*)d_in[7];
    const float* Wc   = (const float*)d_in[8];
    const float* bc   = (const float*)d_in[9];
    const float* Wp   = (const float*)d_in[10];
    const float* bp   = (const float*)d_in[11];
    float* logits = (float*)d_out;

    char* ws = (char*)d_ws;
    size_t off = 0;
    auto carve = [&](size_t bytes) -> void* {
        void* p = ws + off;
        off = (off + bytes + 255) & ~(size_t)255;
        return p;
    };
    float*  seq    = (float*) carve((size_t)CM * CD * 4);       // (T,B,D)
    float*  Xg     = (float*) carve((size_t)CM * G4H * 4);      // (T*B,4H)
    __bf16* W0T    = (__bf16*)carve((size_t)G4H * CH * 2);
    __bf16* W1iT   = (__bf16*)carve((size_t)G4H * CH * 2);
    __bf16* W1hT   = (__bf16*)carve((size_t)G4H * CH * 2);
    __bf16* WpT    = (__bf16*)carve((size_t)CV * CH * 2);       // 32 MB
    float*  h0buf  = (float*) carve((size_t)2 * CB * CH * 4);
    float*  h1buf  = (float*) carve((size_t)2 * CB * CH * 4);
    float*  outb   = (float*) carve((size_t)CM * CH * 4);       // (B,T,H) f32
    __bf16* outbf  = (__bf16*)carve((size_t)CM * CH * 2);       // (B,T,H) bf16
    float*  scores = (float*) carve((size_t)CB * CT * CT * 4);
    float*  wT     = (float*) carve((size_t)CB * CT * CT * 4);
    float*  ctx    = (float*) carve((size_t)CM * CH * 4);
    float*  cat    = (float*) carve((size_t)CM * 2 * CH * 4);
    __bf16* predb  = (__bf16*)carve((size_t)CM * CH * 2);       // bf16 pred_in
    unsigned* ctr  = (unsigned*)carve(256);

    // 1. state/counter init (required every replay)
    init_kernel<<<1, 256, 0, stream>>>(h0buf, h1buf, ctr);
    // 2. embedding gather into (T,B,D)
    embed_kernel<<<dim3(CT, CB), 256, 0, stream>>>(tok, emb, seq);
    // 3. bf16 (N,K) transposes: recurrent weights + Wp
    transpose_bf16_kernel<<<dim3(CD / 32, G4H / 32), 256, 0, stream>>>(Whh0, W0T, CD, G4H);
    transpose_bf16_kernel<<<dim3(CH / 32, G4H / 32), 256, 0, stream>>>(Wih1, W1iT, CH, G4H);
    transpose_bf16_kernel<<<dim3(CH / 32, G4H / 32), 256, 0, stream>>>(Whh1, W1hT, CH, G4H);
    transpose_bf16_kernel<<<dim3(CH / 32, CV / 32), 256, 0, stream>>>(Wp, WpT, CH, CV);
    // 4. Xg = seq @ Wih0 + b0     (M=4096, N=2048, K=512)
    gemm_bf16_kernel<<<dim3(CM / BM, G4H / BN, 1), 256, 0, stream>>>(
        seq, Wih0, b0, Xg, G4H, CD, 0, 0, 0, /*flags=*/4);
    // 5. persistent LSTM recurrence (writes out f32 + bf16)
    lstm_recur_kernel<<<RNB, 128, 0, stream>>>(Xg, W0T, W1iT, W1hT, b1,
                                               h0buf, h1buf, outb, outbf, ctr);
    // 6. scores[b] = out[b] @ out[b]^T  (bf16 x bf16, async staging, batched)
    gemm_bf16x_kernel<<<dim3(CT / BM, CT / BN, CB), 256, 0, stream>>>(
        outbf, outbf, nullptr, scores, CT, CH,
        (long)CT * CH, (long)CT * CH, (long)CT * CT, /*flags=*/0);
    // 7. masked softmax over i, transposed output
    attn_softmax_kernel<<<dim3(CT, CB), 256, 0, stream>>>(scores, tok, wT);
    // 8. ctx[b] = wT[b] @ out[b]        (M=256, N=512, K=256, batched)
    gemm_bf16_kernel<<<dim3(CT / BM, CH / BN, CB), 256, 0, stream>>>(
        wT, outb, nullptr, ctx, CH, CT,
        (long)CT * CT, (long)CT * CH, (long)CT * CH, /*flags=*/0);
    // 9. cat = [ctx | out]
    concat_kernel<<<CM, 256, 0, stream>>>(ctx, outb, cat);
    // 10. predb = bf16(tanh(cat @ Wc + bc))   (M=4096, N=512, K=1024)
    gemm_bf16_kernel<<<dim3(CM / BM, CH / BN, 1), 256, 0, stream>>>(
        cat, Wc, bc, predb, CH, 2 * CH, 0, 0, 0, /*flags=*/4 | 2 | 8);
    // 11. logits = predb @ WpT^T + bp   (bf16 x bf16, async, M=4096, N=32000, K=512)
    gemm_bf16x_kernel<<<dim3(CM / BM, CV / BN, 1), 256, 0, stream>>>(
        predb, WpT, bp, logits, CV, CH, 0, 0, 0, /*flags=*/4);
}
